// DeformAtten2D_45784351375776
// MI455X (gfx1250) — compile-verified, matrix operand order checked by reading, not run on hardware
//
#include <hip/hip_runtime.h>

typedef __bf16 bf16;
typedef __attribute__((ext_vector_type(16))) __bf16 v16bf;
typedef __attribute__((ext_vector_type(8)))  float  v8f;
typedef __attribute__((ext_vector_type(8)))  unsigned int v8u;

#define NPOS 131072          // B*H*W = 8*128*128
#define CCH  256
#define IMG_HW 128

// K-pair base for 16-bit WMMA A/B operand VGPR v (ISA 7.12.2, 16-bit 16x32 layout)
__device__ __forceinline__ int kb_of(int v, int half) {
    return ((v < 4) ? 2 * v : 16 + 2 * (v - 4)) + half * 8;
}

// ---------------------------------------------------------------------------
// bf16 WMMA GEMM, register-blocked: Y[m,n] = sum_k X[m,k]*Wt[n,k] + bias[n]
// (+rel_bias[h(m)]).  N must be 256.  grid = (M/16), block = 128 (4 waves);
// each wave computes a 16x64 slab (4 accumulators), A packed once per k-step.
// ---------------------------------------------------------------------------
__global__ __launch_bounds__(128)
void gemm_bias_wmma(const bf16* __restrict__ X, const bf16* __restrict__ Wt,
                    const float* __restrict__ bias, const float* __restrict__ relbias,
                    float* __restrict__ outF, bf16* __restrict__ outB,
                    int M, int N, int K)
{
    const int lane = threadIdx.x & 31;
    const int wave = threadIdx.x >> 5;
    const int m    = lane & 15;
    const int half = lane >> 4;
    const int m0   = blockIdx.x * 16;
    const int n0   = wave * 64;            // wave covers n0 .. n0+63

    v8f acc[4];
#pragma unroll
    for (int j = 0; j < 4; ++j) {
        const float bv = bias ? bias[n0 + j * 16 + m] : 0.0f;
#pragma unroll
        for (int r = 0; r < 8; ++r) acc[j][r] = bv;
    }

    const bf16* arow = X + (size_t)(m0 + m) * K;
    const bf16* brow[4];
#pragma unroll
    for (int j = 0; j < 4; ++j) brow[j] = Wt + (size_t)(n0 + j * 16 + m) * K;

    for (int k0 = 0; k0 < K; k0 += 32) {
        if (k0 + 32 < K) __builtin_prefetch(arow + k0 + 32, 0, 3);
        v8u au;
#pragma unroll
        for (int v = 0; v < 8; ++v)
            au[v] = *(const unsigned int*)(arow + k0 + kb_of(v, half));
        const v16bf a = __builtin_bit_cast(v16bf, au);
#pragma unroll
        for (int j = 0; j < 4; ++j) {
            v8u bu;
#pragma unroll
            for (int v = 0; v < 8; ++v)
                bu[v] = *(const unsigned int*)(brow[j] + k0 + kb_of(v, half));
            acc[j] = __builtin_amdgcn_wmma_f32_16x16x32_bf16(
                        false, a, false, __builtin_bit_cast(v16bf, bu),
                        (short)0, acc[j], false, false);
        }
    }

    const float rb = relbias ? relbias[((m0 + half * 8) >> 7) & (IMG_HW - 1)] : 0.0f;
#pragma unroll
    for (int j = 0; j < 4; ++j) {
        const int nout = n0 + j * 16 + m;
#pragma unroll
        for (int r = 0; r < 8; ++r) {
            const int row = m0 + half * 8 + r;
            float val = acc[j][r];
            if (relbias) val += relbias[(row >> 7) & (IMG_HW - 1)];
            if (outF) outF[(size_t)row * N + nout] = val;
            if (outB) outB[(size_t)row * N + nout] = (bf16)val;
        }
    }
    (void)rb;
}

// ---------------------------------------------------------------------------
// 3x3 SAME conv as 9 accumulated WMMA GEMMs over per-tap [O,C] weights.
// Same 16x64-per-wave register blocking. X is q [pos,C] bf16, out fp32 [pos,O].
// ---------------------------------------------------------------------------
__global__ __launch_bounds__(128)
void conv3x3_wmma(const bf16* __restrict__ X, const bf16* __restrict__ Wtaps,
                  const float* __restrict__ bias, float* __restrict__ out,
                  int N, int K)
{
    const int lane = threadIdx.x & 31;
    const int wave = threadIdx.x >> 5;
    const int m    = lane & 15;
    const int half = lane >> 4;
    const int m0   = blockIdx.x * 16;
    const int n0   = wave * 64;

    const int row  = m0 + m;              // all 16 rows of the tile share (b,h)
    const int bimg = row >> 14;
    const int h    = (row >> 7) & 127;
    const int w    = row & 127;

    v8f acc[4];
#pragma unroll
    for (int j = 0; j < 4; ++j) {
        const float bv = bias[n0 + j * 16 + m];
#pragma unroll
        for (int r = 0; r < 8; ++r) acc[j][r] = bv;
    }

    const bf16* brow0 = Wtaps + (size_t)(n0 + m) * K;

#pragma unroll
    for (int t = 0; t < 9; ++t) {
        const int dy = t / 3 - 1, dx = t % 3 - 1;
        const int sh = h + dy, sw = w + dx;
        const bool valid = ((unsigned)sh < 128u) && ((unsigned)sw < 128u);
        const size_t spos = valid ? ((size_t)bimg * 16384 + (size_t)sh * 128 + (size_t)sw) : 0;
        const bf16* arow = X + spos * K;
        const bf16* btap = brow0 + (size_t)t * N * K;
        for (int k0 = 0; k0 < K; k0 += 32) {
            v8u au;
#pragma unroll
            for (int v = 0; v < 8; ++v)
                au[v] = valid ? *(const unsigned int*)(arow + k0 + kb_of(v, half)) : 0u;
            const v16bf a = __builtin_bit_cast(v16bf, au);
#pragma unroll
            for (int j = 0; j < 4; ++j) {
                v8u bu;
#pragma unroll
                for (int v = 0; v < 8; ++v)
                    bu[v] = *(const unsigned int*)(btap + (size_t)(j * 16) * K + k0 + kb_of(v, half));
                acc[j] = __builtin_amdgcn_wmma_f32_16x16x32_bf16(
                            false, a, false, __builtin_bit_cast(v16bf, bu),
                            (short)0, acc[j], false, false);
            }
        }
    }

#pragma unroll
    for (int j = 0; j < 4; ++j) {
        const int nout = n0 + j * 16 + m;
#pragma unroll
        for (int r = 0; r < 8; ++r) {
            const int orow = m0 + half * 8 + r;
            out[(size_t)orow * N + nout] = acc[j][r];
        }
    }
}

// ---------------------------------------------------------------------------
// offset = tanh(ho @ W_off2^T) * 3 ; one wave per position, shfl reduction
// ---------------------------------------------------------------------------
__global__ __launch_bounds__(256)
void offset_kernel(const float* __restrict__ ho, const float* __restrict__ Woff2,
                   float* __restrict__ offx, float* __restrict__ offy)
{
    const int lane = threadIdx.x & 31;
    const int wave = threadIdx.x >> 5;
    const size_t pos = (size_t)blockIdx.x * 8 + wave;
    const float* hp = ho + pos * 256;
    float s0 = 0.f, s1 = 0.f;
#pragma unroll
    for (int i = 0; i < 8; ++i) {
        const int c = lane + 32 * i;
        const float hv = hp[c];
        s0 += hv * Woff2[c];
        s1 += hv * Woff2[256 + c];
    }
#pragma unroll
    for (int off = 16; off > 0; off >>= 1) {
        s0 += __shfl_xor(s0, off, 32);
        s1 += __shfl_xor(s1, off, 32);
    }
    if (lane == 0) {
        offx[pos] = tanhf(s0) * 3.0f;
        offy[pos] = tanhf(s1) * 3.0f;
    }
}

// ---------------------------------------------------------------------------
// bilinear grid_sample (zeros padding, align_corners=False torch semantics)
// ix = (w+ox)*W/(W-1) - 0.5 ; block per position, thread per channel
// ---------------------------------------------------------------------------
__global__ __launch_bounds__(256)
void sample_kernel(const float* __restrict__ x, const float* __restrict__ offx,
                   const float* __restrict__ offy, bf16* __restrict__ xs)
{
    const size_t pos = blockIdx.x;
    const int c    = threadIdx.x;
    const int bimg = (int)(pos >> 14);
    const int h    = (int)((pos >> 7) & 127);
    const int w    = (int)(pos & 127);
    const float sc = 128.0f / 127.0f;
    const float ix = ((float)w + offx[pos]) * sc - 0.5f;
    const float iy = ((float)h + offy[pos]) * sc - 0.5f;
    const float fx0 = floorf(ix), fy0 = floorf(iy);
    const int ix0 = (int)fx0, iy0 = (int)fy0;
    const float wx1 = ix - fx0, wx0 = 1.0f - wx1;
    const float wy1 = iy - fy0, wy0 = 1.0f - wy1;
    const float* xb = x + (size_t)bimg * 16384 * 256;
    float acc = 0.f;
#pragma unroll
    for (int j = 0; j < 4; ++j) {
        const int yi = iy0 + (j >> 1), xi = ix0 + (j & 1);
        const float wgt = ((j >> 1) ? wy1 : wy0) * ((j & 1) ? wx1 : wx0);
        if ((unsigned)yi < 128u && (unsigned)xi < 128u)
            acc += wgt * xb[((size_t)yi * 128 + xi) * 256 + c];
    }
    xs[pos * 256 + c] = (bf16)acc;
}

// ---------------------------------------------------------------------------
// [pos, C] -> head-major [(b*C+c), h, w] so each head is a contiguous 128x128
// ---------------------------------------------------------------------------
__global__ __launch_bounds__(256)
void head_transpose(const bf16* __restrict__ src, bf16* __restrict__ dst)
{
    const size_t pos = blockIdx.x;
    const int c = threadIdx.x;
    const size_t bimg = pos >> 14;
    const size_t rem  = pos & 16383;
    dst[((bimg * 256 + (size_t)c) << 14) + rem] = src[pos * 256 + c];
}

// ---------------------------------------------------------------------------
// Per-head attention: S = QK^T/16 (WMMA) -> softmax in LDS -> O = PV (WMMA)
// grid = 2048 heads, block = 128 (4 waves). 64KB LDS for S/P.
// ---------------------------------------------------------------------------
__global__ __launch_bounds__(128)
void attn_kernel(const bf16* __restrict__ qT, const bf16* __restrict__ kT,
                 const bf16* __restrict__ vT, bf16* __restrict__ yb)
{
    __shared__ float Sbuf[128 * 128];    // 64 KB
    const int lane = threadIdx.x & 31;
    const int wave = threadIdx.x >> 5;
    const int m    = lane & 15;
    const int half = lane >> 4;
    const size_t base = (size_t)blockIdx.x << 14;
    const bf16* Q  = qT + base;
    const bf16* Kh = kT + base;
    const unsigned short* Vh = (const unsigned short*)(vT + base);

    // Phase 1: S = Q K^T * 1/sqrt(C)
    for (int t = wave; t < 64; t += 4) {
        const int i0 = (t >> 3) * 16, j0 = (t & 7) * 16;
        v8f acc;
#pragma unroll
        for (int r = 0; r < 8; ++r) acc[r] = 0.f;
        const bf16* arow = Q  + (size_t)(i0 + m) * 128;
        const bf16* brow = Kh + (size_t)(j0 + m) * 128;
#pragma unroll
        for (int d0 = 0; d0 < 128; d0 += 32) {
            v8u au, bu;
#pragma unroll
            for (int v = 0; v < 8; ++v) {
                const int kb = kb_of(v, half);
                au[v] = *(const unsigned int*)(arow + d0 + kb);
                bu[v] = *(const unsigned int*)(brow + d0 + kb);
            }
            acc = __builtin_amdgcn_wmma_f32_16x16x32_bf16(
                    false, __builtin_bit_cast(v16bf, au),
                    false, __builtin_bit_cast(v16bf, bu),
                    (short)0, acc, false, false);
        }
#pragma unroll
        for (int r = 0; r < 8; ++r)
            Sbuf[(i0 + half * 8 + r) * 128 + j0 + m] = acc[r] * 0.0625f;
    }
    __syncthreads();

    // Phase 2: row softmax (wave-per-row, 4 elems/lane)
    for (int row = wave; row < 128; row += 4) {
        float e[4];
        float mx = -3.4e38f;
#pragma unroll
        for (int i = 0; i < 4; ++i) { e[i] = Sbuf[row * 128 + lane + 32 * i]; mx = fmaxf(mx, e[i]); }
#pragma unroll
        for (int off = 16; off > 0; off >>= 1) mx = fmaxf(mx, __shfl_xor(mx, off, 32));
        float s = 0.f;
#pragma unroll
        for (int i = 0; i < 4; ++i) { e[i] = __expf(e[i] - mx); s += e[i]; }
#pragma unroll
        for (int off = 16; off > 0; off >>= 1) s += __shfl_xor(s, off, 32);
        const float inv = 1.0f / s;
#pragma unroll
        for (int i = 0; i < 4; ++i) Sbuf[row * 128 + lane + 32 * i] = e[i] * inv;
    }
    __syncthreads();

    // Phase 3: O = P V
    for (int t = wave; t < 64; t += 4) {
        const int i0 = (t >> 3) * 16, d0 = (t & 7) * 16;
        v8f acc;
#pragma unroll
        for (int r = 0; r < 8; ++r) acc[r] = 0.f;
#pragma unroll
        for (int j0 = 0; j0 < 128; j0 += 32) {
            v16bf a; v8u bu;
#pragma unroll
            for (int v = 0; v < 8; ++v) {
                const int kb = kb_of(v, half);
                a[2 * v]     = (bf16)Sbuf[(i0 + m) * 128 + j0 + kb];
                a[2 * v + 1] = (bf16)Sbuf[(i0 + m) * 128 + j0 + kb + 1];
                const unsigned lo = Vh[(j0 + kb)     * 128 + d0 + m];
                const unsigned hi = Vh[(j0 + kb + 1) * 128 + d0 + m];
                bu[v] = lo | (hi << 16);
            }
            acc = __builtin_amdgcn_wmma_f32_16x16x32_bf16(
                    false, a,
                    false, __builtin_bit_cast(v16bf, bu),
                    (short)0, acc, false, false);
        }
#pragma unroll
        for (int r = 0; r < 8; ++r)
            yb[base + (size_t)(i0 + half * 8 + r) * 128 + d0 + m] = (bf16)acc[r];
    }
}

// ---------------------------------------------------------------------------
// conversion helpers
// ---------------------------------------------------------------------------
__global__ void f2b_kernel(const float* __restrict__ src, bf16* __restrict__ dst, size_t n)
{
    size_t i = (size_t)blockIdx.x * blockDim.x + threadIdx.x;
    const size_t stride = (size_t)gridDim.x * blockDim.x;
    for (; i < n; i += stride) dst[i] = (bf16)src[i];
}

// W_off1 [O,C,3,3] -> 9 per-tap [O,C] bf16 matrices
__global__ void taps_f2b_kernel(const float* __restrict__ w, bf16* __restrict__ dst)
{
    const int i = blockIdx.x * blockDim.x + threadIdx.x;
    if (i >= 9 * 256 * 256) return;
    const int t = i >> 16;
    const int o = (i >> 8) & 255;
    const int c = i & 255;
    dst[i] = (bf16)w[(size_t)o * 2304 + c * 9 + t];
}

// ---------------------------------------------------------------------------
extern "C" void kernel_launch(void* const* d_in, const int* in_sizes, int n_in,
                              void* d_out, int out_size, void* d_ws, size_t ws_size,
                              hipStream_t stream)
{
    (void)in_sizes; (void)n_in; (void)out_size; (void)ws_size;
    const float* x     = (const float*)d_in[0];
    const float* Wq    = (const float*)d_in[1];
    const float* bq    = (const float*)d_in[2];
    const float* Wk    = (const float*)d_in[3];
    const float* bk    = (const float*)d_in[4];
    const float* Wv    = (const float*)d_in[5];
    const float* bv    = (const float*)d_in[6];
    const float* Woff1 = (const float*)d_in[7];
    const float* boff1 = (const float*)d_in[8];
    const float* Woff2 = (const float*)d_in[9];
    const float* relb  = (const float*)d_in[10];
    const float* Wout  = (const float*)d_in[11];
    const float* bout  = (const float*)d_in[12];

    char* ws = (char*)d_ws;
    const size_t MB = 1ull << 20;
    bf16*  xb    = (bf16*)(ws + 0);            // 64 MiB: xb -> xs -> qT
    bf16*  qb    = (bf16*)(ws + 64 * MB);      // 64 MiB: q  -> yb
    float* ho    = (float*)(ws + 128 * MB);    // 128 MiB: ho -> k_b / v_b
    bf16*  kb_   = (bf16*)(ws + 128 * MB);
    bf16*  vb_   = (bf16*)(ws + 192 * MB);
    bf16*  kT    = (bf16*)(ws + 256 * MB);     // 64 MiB
    bf16*  vT    = (bf16*)(ws + 320 * MB);     // 64 MiB
    float* offx  = (float*)(ws + 384 * MB);
    float* offy  = (float*)(ws + 384 * MB + 512 * 1024);
    bf16*  wqb   = (bf16*)(ws + 385 * MB);
    bf16*  wkb   = (bf16*)(ws + 385 * MB + 128 * 1024);
    bf16*  wvb   = (bf16*)(ws + 385 * MB + 256 * 1024);
    bf16*  wob   = (bf16*)(ws + 385 * MB + 384 * 1024);
    bf16*  wtaps = (bf16*)(ws + 385 * MB + 512 * 1024);
    bf16*  xs = xb;      // alias after q-GEMM consumed xb
    bf16*  qT = xb;      // alias after k/v GEMMs consumed xs
    bf16*  yb = qb;      // alias after q transposed

    const size_t NEL = (size_t)NPOS * CCH;

    f2b_kernel<<<4096, 256, 0, stream>>>(x, xb, NEL);
    f2b_kernel<<<256, 256, 0, stream>>>(Wq, wqb, 65536);
    f2b_kernel<<<256, 256, 0, stream>>>(Wk, wkb, 65536);
    f2b_kernel<<<256, 256, 0, stream>>>(Wv, wvb, 65536);
    f2b_kernel<<<256, 256, 0, stream>>>(Wout, wob, 65536);
    taps_f2b_kernel<<<(9 * 256 * 256 + 255) / 256, 256, 0, stream>>>(Woff1, wtaps);

    dim3 gg(NPOS / 16);   // one 16-row M-tile per block; block covers all N=256
    // q = x Wq^T + bq
    gemm_bias_wmma<<<gg, 128, 0, stream>>>(xb, wqb, bq, nullptr, nullptr, qb, NPOS, 256, 256);
    // ho = conv3x3(q) + b_off1
    conv3x3_wmma<<<gg, 128, 0, stream>>>(qb, wtaps, boff1, ho, 256, 256);
    // offsets
    offset_kernel<<<NPOS / 8, 256, 0, stream>>>(ho, Woff2, offx, offy);
    // x_sampled
    sample_kernel<<<NPOS, 256, 0, stream>>>(x, offx, offy, xs);
    // k, v
    gemm_bias_wmma<<<gg, 128, 0, stream>>>(xs, wkb, bk, nullptr, nullptr, kb_, NPOS, 256, 256);
    gemm_bias_wmma<<<gg, 128, 0, stream>>>(xs, wvb, bv, relb, nullptr, vb_, NPOS, 256, 256);
    // head-major transposes
    head_transpose<<<NPOS, 256, 0, stream>>>(qb, qT);
    head_transpose<<<NPOS, 256, 0, stream>>>(kb_, kT);
    head_transpose<<<NPOS, 256, 0, stream>>>(vb_, vT);
    // attention (2048 heads)
    attn_kernel<<<2048, 128, 0, stream>>>(qT, kT, vT, yb);
    // final projection -> d_out (fp32)
    gemm_bias_wmma<<<gg, 128, 0, stream>>>(yb, wob, bout, nullptr, (float*)d_out, nullptr, NPOS, 256, 256);
}
